// Decoder_16922171147076
// MI455X (gfx1250) — matrix operand order repaired and checked by reading, not compiled, not run on hardware
//
#include <hip/hip_runtime.h>

// MI455X (gfx1250) fused recurrent decoder.
// One workgroup (256 threads = 8 wave32) per scene of 64 agents; all 12
// recurrent steps run in-kernel with state + weights resident in LDS.
// All GEMMs via v_wmma_f32_16x16x32_f16 (f16 in, f32 accumulate).
// Nonlinearities use v_rcp_f32 (1-ulp) instead of IEEE division sequences.

#define OBS_LEN 8
#define PRED_LEN 12
#define BTOT 32768
#define NGRP 512
#define GAG 64      // agents per scene
#define GHD 64      // goal hidden
#define AHD 32      // action hidden

typedef __attribute__((ext_vector_type(16))) _Float16 v16h;
typedef __attribute__((ext_vector_type(8)))  float    v8f;

__device__ __forceinline__ float frcp(float x) { return __builtin_amdgcn_rcpf(x); }
__device__ __forceinline__ float sigf(float x) { return frcp(1.0f + __expf(-x)); }
__device__ __forceinline__ float tanhfast(float x) {
  x = fminf(fmaxf(x, -15.0f), 15.0f);
  float e = __expf(2.0f * x);
  return (e - 1.0f) * frcp(e + 1.0f);
}
__device__ __forceinline__ float eluf(float x) { return x > 0.0f ? x : (__expf(x) - 1.0f); }

// ---- WMMA fragment loaders (CDNA5 ISA 7.12.2 layouts) ----
// A: 16x32 f16, row-major source [M][lda]. lane<16 holds K {0..7,16..23},
// lane>=16 holds K {8..15,24..31}; row = row0 + (lane&15).
__device__ __forceinline__ v16h load_A_f16(const _Float16* A, int lda, int row0, int k0, int lane) {
  const _Float16* p = A + (row0 + (lane & 15)) * lda + k0;
  const int kh = (lane >> 4) << 3;   // 0 or 8
  v16h a;
#pragma unroll
  for (int i = 0; i < 8; ++i) { a[i] = p[kh + i]; a[8 + i] = p[16 + kh + i]; }
  return a;
}
// B: 32x16 f16, stored transposed as BT[N][ldb] (BT[n][k] = B[k][n]).
// lane column n = n0 + (lane&15); lane<16 K {0..7,16..23}, lane>=16 K {8..15,24..31}.
__device__ __forceinline__ v16h load_BT_f16(const _Float16* BT, int ldb, int n0, int k0, int lane) {
  const _Float16* p = BT + (n0 + (lane & 15)) * ldb + k0 + ((lane >> 4) << 3);
  v16h b;
#pragma unroll
  for (int i = 0; i < 8; ++i) { b[i] = p[i]; b[8 + i] = p[16 + i]; }
  return b;
}
__device__ __forceinline__ v8f wmma16(v16h a, v16h b, v8f c) {
  return __builtin_amdgcn_wmma_f32_16x16x32_f16(false, a, false, b, (short)0, c, false, false);
}

struct __align__(16) Smem {
  // persistent per-scene state (f32)
  float gh[GAG][GHD];
  float gc[GAG][GHD];
  float ah[GAG][AHD];
  float ac[GAG][AHD];
  float go[GAG][2];
  float ao[GAG][2];
  // WMMA B operands (f16, stored [N][K])
  _Float16 whhg[4 * GHD][GHD];   // goal  LSTM Whh  [256][64]
  _Float16 whha[4 * AHD][AHD];   // action LSTM Whh [128][32]
  _Float16 w0T[64][32];          // GAT0: [head*16+o][k]
  _Float16 w1T[32][64];          // GAT1: [o][k]
  // small f32 params
  float wihg[4 * GHD][2]; float bg[4 * GHD];
  float wiha[4 * AHD][2]; float ba[4 * AHD];
  float wh2g[2][GHD];     float bh2g[2];
  float wh2a[2][AHD];     float bh2a[2];
  float wga[AHD][2];      float bga[AHD];
  float waa[GHD][2];      float baa[GHD];
  float as0[4][16]; float ad0[4][16]; float b0v[16];
  float as1[32];    float ad1[32];    float b1v[32];
  // transient
  _Float16 A16g[GAG][GHD];   // f16 A operand, 64x64
  _Float16 A16a[GAG][AHD];   // f16 A operand, 64x32
  _Float16 hpT[64][GAG];     // GAT0 projections, transposed [feat][agent]
  _Float16 attnU[GAG][GAG];  // unnormalized attn exp(x-max)
  _Float16 yT[64][GAG];      // GAT0 output (elu), transposed
  _Float16 hp1T[32][GAG];    // GAT1 projections, transposed
  float sArr[GAG]; float dArr[GAG]; float rsArr[GAG];
  float meanv[64]; float invv[64];
};

__global__ void __launch_bounds__(256)
decoder_scene_kernel(
    const float* __restrict__ goal_real,  const float* __restrict__ gih,
    const float* __restrict__ action_real,const float* __restrict__ aih,
    const float* __restrict__ Wih_g, const float* __restrict__ Whh_g,
    const float* __restrict__ bih_g, const float* __restrict__ bhh_g,
    const float* __restrict__ W_h2g, const float* __restrict__ b_h2g,
    const float* __restrict__ Wih_a, const float* __restrict__ Whh_a,
    const float* __restrict__ bih_a, const float* __restrict__ bhh_a,
    const float* __restrict__ W_h2a, const float* __restrict__ b_h2a,
    const float* __restrict__ W_ga,  const float* __restrict__ b_ga,
    const float* __restrict__ W_aa,  const float* __restrict__ b_aa,
    const float* __restrict__ w0,    const float* __restrict__ asrc0,
    const float* __restrict__ adst0, const float* __restrict__ bias0,
    const float* __restrict__ w1,    const float* __restrict__ asrc1,
    const float* __restrict__ adst1, const float* __restrict__ bias1,
    float* __restrict__ out)
{
  __shared__ Smem sm;
  const int tid  = threadIdx.x;
  const int wid  = tid >> 5;
  const int lane = tid & 31;
  const int n15  = lane & 15;
  const int rb   = (lane >> 4) << 3;
  const int scene = blockIdx.x;
  const int ped0  = scene * GAG;

  // ---------------- init: state + weights -> LDS ----------------
  for (int i = tid; i < GAG * GHD; i += 256) {
    ((float*)sm.gh)[i] = gih[(size_t)ped0 * GHD + i];
    ((float*)sm.gc)[i] = 0.0f;
  }
  for (int i = tid; i < GAG * AHD; i += 256) {
    ((float*)sm.ah)[i] = aih[(size_t)ped0 * AHD + i];
    ((float*)sm.ac)[i] = 0.0f;
  }
  for (int i = tid; i < GAG * 2; i += 256) {
    const size_t base = (size_t)(OBS_LEN - 1) * BTOT * 2 + (size_t)ped0 * 2;
    ((float*)sm.go)[i] = goal_real[base + i];
    ((float*)sm.ao)[i] = action_real[base + i];
  }
  for (int i = tid; i < 4 * GHD * GHD; i += 256) ((_Float16*)sm.whhg)[i] = (_Float16)Whh_g[i];
  for (int i = tid; i < 4 * AHD * AHD; i += 256) ((_Float16*)sm.whha)[i] = (_Float16)Whh_a[i];
  for (int i = tid; i < 64 * 32; i += 256) {            // w0 [4,32,16] -> w0T[h*16+o][k]
    int f = i >> 5, k = i & 31, h = f >> 4, o = f & 15;
    sm.w0T[f][k] = (_Float16)w0[(h * 32 + k) * 16 + o];
  }
  for (int i = tid; i < 32 * 64; i += 256) {            // w1 [1,64,32] -> w1T[o][k]
    int n = i >> 6, k = i & 63;
    sm.w1T[n][k] = (_Float16)w1[k * 32 + n];
  }
  for (int i = tid; i < 4 * GHD * 2; i += 256) ((float*)sm.wihg)[i] = Wih_g[i];
  for (int i = tid; i < 4 * GHD; i += 256) sm.bg[i] = bih_g[i] + bhh_g[i];
  for (int i = tid; i < 4 * AHD * 2; i += 256) ((float*)sm.wiha)[i] = Wih_a[i];
  for (int i = tid; i < 4 * AHD; i += 256) sm.ba[i] = bih_a[i] + bhh_a[i];
  for (int i = tid; i < 2 * GHD; i += 256) ((float*)sm.wh2g)[i] = W_h2g[i];
  for (int i = tid; i < 2 * AHD; i += 256) ((float*)sm.wh2a)[i] = W_h2a[i];
  for (int i = tid; i < AHD * 2; i += 256) ((float*)sm.wga)[i] = W_ga[i];
  for (int i = tid; i < GHD * 2; i += 256) ((float*)sm.waa)[i] = W_aa[i];
  if (tid < 2)  { sm.bh2g[tid] = b_h2g[tid]; sm.bh2a[tid] = b_h2a[tid]; }
  if (tid < AHD) sm.bga[tid] = b_ga[tid];
  if (tid < GHD) sm.baa[tid] = b_aa[tid];
  if (tid < 64)  { ((float*)sm.as0)[tid] = asrc0[tid]; ((float*)sm.ad0)[tid] = adst0[tid]; }
  if (tid < 16)  sm.b0v[tid] = bias0[tid];
  if (tid < 32)  { sm.as1[tid] = asrc1[tid]; sm.ad1[tid] = adst1[tid]; sm.b1v[tid] = bias1[tid]; }
  __syncthreads();

  // ---------------- 12 recurrent steps ----------------
  for (int t = 0; t < PRED_LEN; ++t) {
    // A) stage f16 copies of h states for WMMA A operands
    for (int i = tid; i < GAG * GHD; i += 256) ((_Float16*)sm.A16g)[i] = (_Float16)((float*)sm.gh)[i];
    for (int i = tid; i < GAG * AHD; i += 256) ((_Float16*)sm.A16a)[i] = (_Float16)((float*)sm.ah)[i];
    __syncthreads();

    // B) waves 0-3: goal LSTM (64x64 -> 256); waves 4-7: action LSTM (64x32 -> 128)
    if (wid < 4) {
      const int m0 = wid * 16;
      v16h a0 = load_A_f16(&sm.A16g[0][0], GHD, m0, 0, lane);
      v16h a1 = load_A_f16(&sm.A16g[0][0], GHD, m0, 32, lane);
      float x0[8], x1[8];
#pragma unroll
      for (int r = 0; r < 8; ++r) { int m = m0 + rb + r; x0[r] = sm.go[m][0]; x1[r] = sm.go[m][1]; }
#pragma unroll
      for (int q = 0; q < 4; ++q) {
        v8f z[4];
#pragma unroll
        for (int g4 = 0; g4 < 4; ++g4) {               // gates i,f,g,o
          const int nt = g4 * 4 + q, n = nt * 16 + n15;
          const float wg0 = sm.wihg[n][0], wg1 = sm.wihg[n][1], bb = sm.bg[n];
          v8f acc;
#pragma unroll
          for (int r = 0; r < 8; ++r) acc[r] = x0[r] * wg0 + x1[r] * wg1 + bb;
          acc = wmma16(a0, load_BT_f16(&sm.whhg[0][0], GHD, nt * 16, 0, lane), acc);
          acc = wmma16(a1, load_BT_f16(&sm.whhg[0][0], GHD, nt * 16, 32, lane), acc);
          z[g4] = acc;
        }
        const int j = q * 16 + n15;
#pragma unroll
        for (int r = 0; r < 8; ++r) {
          const int m = m0 + rb + r;
          float c = sigf(z[1][r]) * sm.gc[m][j] + sigf(z[0][r]) * tanhfast(z[2][r]);
          sm.gc[m][j] = c;
          sm.gh[m][j] = sigf(z[3][r]) * tanhfast(c);
        }
      }
    } else {
      const int m0 = (wid - 4) * 16;
      v16h a0 = load_A_f16(&sm.A16a[0][0], AHD, m0, 0, lane);
      float x0[8], x1[8];
#pragma unroll
      for (int r = 0; r < 8; ++r) { int m = m0 + rb + r; x0[r] = sm.ao[m][0]; x1[r] = sm.ao[m][1]; }
#pragma unroll
      for (int q = 0; q < 2; ++q) {
        v8f z[4];
#pragma unroll
        for (int g4 = 0; g4 < 4; ++g4) {
          const int nt = g4 * 2 + q, n = nt * 16 + n15;
          const float wg0 = sm.wiha[n][0], wg1 = sm.wiha[n][1], bb = sm.ba[n];
          v8f acc;
#pragma unroll
          for (int r = 0; r < 8; ++r) acc[r] = x0[r] * wg0 + x1[r] * wg1 + bb;
          z[g4] = wmma16(a0, load_BT_f16(&sm.whha[0][0], AHD, nt * 16, 0, lane), acc);
        }
        const int j = q * 16 + n15;
#pragma unroll
        for (int r = 0; r < 8; ++r) {
          const int m = m0 + rb + r;
          float c = sigf(z[1][r]) * sm.ac[m][j] + sigf(z[0][r]) * tanhfast(z[2][r]);
          sm.ac[m][j] = c;
          sm.ah[m][j] = sigf(z[3][r]) * tanhfast(c);
        }
      }
    }
    __syncthreads();

    // C) goal_out = gh @ W_h2g^T + b ; emit pred_goal
    if (tid < 128) {
      const int m = tid >> 1, c = tid & 1;
      float s = sm.bh2g[c];
      for (int k = 0; k < GHD; ++k) s += sm.gh[m][k] * sm.wh2g[c][k];
      sm.go[m][c] = s;
      out[(size_t)t * BTOT * 2 + (size_t)(ped0 + m) * 2 + c] = s;
    }
    __syncthreads();

    // D) GoalAttention: ah *= softmax(go @ W_ga^T + b_ga) over 32 channels
    if (tid < GAG) {
      const int m = tid;
      const float g0 = sm.go[m][0], g1 = sm.go[m][1];
      float mx = -1e30f;
      for (int j = 0; j < AHD; ++j)
        mx = fmaxf(mx, g0 * sm.wga[j][0] + g1 * sm.wga[j][1] + sm.bga[j]);
      float Z = 0.0f;
      for (int j = 0; j < AHD; ++j)
        Z += __expf(g0 * sm.wga[j][0] + g1 * sm.wga[j][1] + sm.bga[j] - mx);
      const float iz = frcp(Z);
      for (int j = 0; j < AHD; ++j)
        sm.ah[m][j] *= __expf(g0 * sm.wga[j][0] + g1 * sm.wga[j][1] + sm.bga[j] - mx) * iz;
    }
    __syncthreads();

    // E) InstanceNorm over agents (per channel) + stage f16 A
    if (tid < AHD) {
      const int ch = tid; float s = 0.0f, s2 = 0.0f;
      for (int m = 0; m < GAG; ++m) { float v = sm.ah[m][ch]; s += v; s2 += v * v; }
      const float mu = s * (1.0f / GAG);
      sm.meanv[ch] = mu;
      sm.invv[ch]  = rsqrtf(s2 * (1.0f / GAG) - mu * mu + 1e-5f);
    }
    __syncthreads();
    for (int i = tid; i < GAG * AHD; i += 256) {
      const int m = i / AHD, ch = i % AHD;
      sm.A16a[m][ch] = (_Float16)((sm.ah[m][ch] - sm.meanv[ch]) * sm.invv[ch]);
    }
    __syncthreads();

    // F) GAT0 projection: [64x32] @ w0 -> hpT [64feat][64agent]
    for (int it = wid; it < 16; it += 8) {
      const int mt = it >> 2, ntile = it & 3;
      v16h a = load_A_f16(&sm.A16a[0][0], AHD, mt * 16, 0, lane);
      v16h b = load_BT_f16(&sm.w0T[0][0], 32, ntile * 16, 0, lane);
      v8f acc = {0.f,0.f,0.f,0.f,0.f,0.f,0.f,0.f};
      acc = wmma16(a, b, acc);
      const int n = ntile * 16 + n15;
#pragma unroll
      for (int r = 0; r < 8; ++r) sm.hpT[n][mt * 16 + rb + r] = (_Float16)acc[r];
    }
    __syncthreads();

    // G) GAT0 attention, one head at a time
    for (int h = 0; h < 4; ++h) {
      if (tid < 128) {                                  // s,d dots (16-wide)
        const int n = tid & 63;
        const float* av = (tid < 64) ? sm.as0[h] : sm.ad0[h];
        float s = 0.0f;
        for (int o = 0; o < 16; ++o) s += (float)sm.hpT[h * 16 + o][n] * av[o];
        if (tid < 64) sm.sArr[n] = s; else sm.dArr[n] = s;
      }
      __syncthreads();
      if (tid < GAG) {                                  // row softmax (unnormalized)
        const int n = tid; const float sn = sm.sArr[n];
        float mx = -1e30f;
        for (int m = 0; m < GAG; ++m) {
          float v = sn + sm.dArr[m]; v = v > 0.0f ? v : 0.2f * v;
          mx = fmaxf(mx, v);
        }
        float Z = 0.0f;
        for (int m = 0; m < GAG; ++m) {
          float v = sn + sm.dArr[m]; v = v > 0.0f ? v : 0.2f * v;
          float e = __expf(v - mx); Z += e;
          sm.attnU[n][m] = (_Float16)e;
        }
        sm.rsArr[n] = frcp(Z);
      }
      __syncthreads();
      if (wid < 4) {                                    // attn @ hp (64x64x16)
        const int mt = wid;
        v16h a0 = load_A_f16(&sm.attnU[0][0], GAG, mt * 16, 0, lane);
        v16h a1 = load_A_f16(&sm.attnU[0][0], GAG, mt * 16, 32, lane);
        v16h b0 = load_BT_f16(&sm.hpT[h * 16][0], GAG, 0, 0, lane);
        v16h b1 = load_BT_f16(&sm.hpT[h * 16][0], GAG, 0, 32, lane);
        v8f acc = {0.f,0.f,0.f,0.f,0.f,0.f,0.f,0.f};
        acc = wmma16(a0, b0, acc);
        acc = wmma16(a1, b1, acc);
        const float bb = sm.b0v[n15];
#pragma unroll
        for (int r = 0; r < 8; ++r) {
          const int m = mt * 16 + rb + r;
          float v = acc[r] * sm.rsArr[m] + bb;          // normalize + bias
          sm.yT[h * 16 + n15][m] = (_Float16)eluf(v);   // elu between layers
        }
      }
      __syncthreads();
    }

    // H) GAT1: norm -> proj -> attention -> new ah
    if (tid < 64) {
      const int ch = tid; float s = 0.0f, s2 = 0.0f;
      for (int m = 0; m < GAG; ++m) { float v = (float)sm.yT[ch][m]; s += v; s2 += v * v; }
      const float mu = s * (1.0f / GAG);
      sm.meanv[ch] = mu;
      sm.invv[ch]  = rsqrtf(s2 * (1.0f / GAG) - mu * mu + 1e-5f);
    }
    __syncthreads();
    for (int i = tid; i < GAG * 64; i += 256) {
      const int m = i >> 6, ch = i & 63;
      sm.A16g[m][ch] = (_Float16)(((float)sm.yT[ch][m] - sm.meanv[ch]) * sm.invv[ch]);
    }
    __syncthreads();
    {                                                   // proj: 64x64 -> 32, 8 tiles
      const int mt = wid >> 1, ntile = wid & 1;
      v16h a0 = load_A_f16(&sm.A16g[0][0], GHD, mt * 16, 0, lane);
      v16h a1 = load_A_f16(&sm.A16g[0][0], GHD, mt * 16, 32, lane);
      v16h b0 = load_BT_f16(&sm.w1T[0][0], 64, ntile * 16, 0, lane);
      v16h b1 = load_BT_f16(&sm.w1T[0][0], 64, ntile * 16, 32, lane);
      v8f acc = {0.f,0.f,0.f,0.f,0.f,0.f,0.f,0.f};
      acc = wmma16(a0, b0, acc);
      acc = wmma16(a1, b1, acc);
      const int n = ntile * 16 + n15;
#pragma unroll
      for (int r = 0; r < 8; ++r) sm.hp1T[n][mt * 16 + rb + r] = (_Float16)acc[r];
    }
    __syncthreads();
    if (tid < 128) {                                    // s1,d1 dots (32-wide)
      const int n = tid & 63;
      const float* av = (tid < 64) ? sm.as1 : sm.ad1;
      float s = 0.0f;
      for (int o = 0; o < 32; ++o) s += (float)sm.hp1T[o][n] * av[o];
      if (tid < 64) sm.sArr[n] = s; else sm.dArr[n] = s;
    }
    __syncthreads();
    if (tid < GAG) {
      const int n = tid; const float sn = sm.sArr[n];
      float mx = -1e30f;
      for (int m = 0; m < GAG; ++m) {
        float v = sn + sm.dArr[m]; v = v > 0.0f ? v : 0.2f * v;
        mx = fmaxf(mx, v);
      }
      float Z = 0.0f;
      for (int m = 0; m < GAG; ++m) {
        float v = sn + sm.dArr[m]; v = v > 0.0f ? v : 0.2f * v;
        float e = __expf(v - mx); Z += e;
        sm.attnU[n][m] = (_Float16)e;
      }
      sm.rsArr[n] = frcp(Z);
    }
    __syncthreads();
    {                                                   // attn1 @ hp1 -> ah (64x64x32)
      const int mt = wid >> 1, ntile = wid & 1;
      v16h a0 = load_A_f16(&sm.attnU[0][0], GAG, mt * 16, 0, lane);
      v16h a1 = load_A_f16(&sm.attnU[0][0], GAG, mt * 16, 32, lane);
      v16h b0 = load_BT_f16(&sm.hp1T[0][0], GAG, ntile * 16, 0, lane);
      v16h b1 = load_BT_f16(&sm.hp1T[0][0], GAG, ntile * 16, 32, lane);
      v8f acc = {0.f,0.f,0.f,0.f,0.f,0.f,0.f,0.f};
      acc = wmma16(a0, b0, acc);
      acc = wmma16(a1, b1, acc);
      const int n = ntile * 16 + n15;
      const float bb = sm.b1v[n];
#pragma unroll
      for (int r = 0; r < 8; ++r) {
        const int m = mt * 16 + rb + r;
        sm.ah[m][n] = acc[r] * sm.rsArr[m] + bb;
      }
    }
    __syncthreads();

    // I) action_out = ah @ W_h2a^T + b ; emit pred_action
    if (tid < 128) {
      const int m = tid >> 1, c = tid & 1;
      float s = sm.bh2a[c];
      for (int k = 0; k < AHD; ++k) s += sm.ah[m][k] * sm.wh2a[c][k];
      sm.ao[m][c] = s;
      out[(size_t)PRED_LEN * BTOT * 2 + (size_t)t * BTOT * 2 + (size_t)(ped0 + m) * 2 + c] = s;
    }
    __syncthreads();

    // J) ActionAttention: gh *= softmax(ao @ W_aa^T + b_aa) over 64 channels
    if (tid < GAG) {
      const int m = tid;
      const float a0v = sm.ao[m][0], a1v = sm.ao[m][1];
      float mx = -1e30f;
      for (int j = 0; j < GHD; ++j)
        mx = fmaxf(mx, a0v * sm.waa[j][0] + a1v * sm.waa[j][1] + sm.baa[j]);
      float Z = 0.0f;
      for (int j = 0; j < GHD; ++j)
        Z += __expf(a0v * sm.waa[j][0] + a1v * sm.waa[j][1] + sm.baa[j] - mx);
      const float iz = frcp(Z);
      for (int j = 0; j < GHD; ++j)
        sm.gh[m][j] *= __expf(a0v * sm.waa[j][0] + a1v * sm.waa[j][1] + sm.baa[j] - mx) * iz;
    }
    __syncthreads();
  }
}

extern "C" void kernel_launch(void* const* d_in, const int* in_sizes, int n_in,
                              void* d_out, int out_size, void* d_ws, size_t ws_size,
                              hipStream_t stream) {
  (void)in_sizes; (void)n_in; (void)out_size; (void)d_ws; (void)ws_size;
  decoder_scene_kernel<<<NGRP, 256, 0, stream>>>(
      (const float*)d_in[2],  (const float*)d_in[3],
      (const float*)d_in[4],  (const float*)d_in[5],
      (const float*)d_in[6],  (const float*)d_in[7],
      (const float*)d_in[8],  (const float*)d_in[9],
      (const float*)d_in[10], (const float*)d_in[11],
      (const float*)d_in[12], (const float*)d_in[13],
      (const float*)d_in[14], (const float*)d_in[15],
      (const float*)d_in[16], (const float*)d_in[17],
      (const float*)d_in[18], (const float*)d_in[19],
      (const float*)d_in[20], (const float*)d_in[21],
      (const float*)d_in[22], (const float*)d_in[23],
      (const float*)d_in[24], (const float*)d_in[25],
      (const float*)d_in[26], (const float*)d_in[27],
      (const float*)d_in[28], (const float*)d_in[29],
      (float*)d_out);
}
